// MixtralSparseMoe_8400956031007
// MI455X (gfx1250) — matrix operand order repaired and checked by reading, not compiled, not run on hardware
//
#include <hip/hip_runtime.h>
#include <hip/hip_bf16.h>

// ---------------------------------------------------------------------------
// Mixtral sparse MoE for gfx1250 (MI455X, wave32, WMMA bf16, async-to-LDS).
//
// Shapes: B=4 S=2048 -> T=8192 tokens, D=1024, F=4096, E=8, top_k=2.
// d_out = [ out: T*D f32 | router_logits: T*E f32 ]
//
// Stages (all on `stream`):
//   0) zero d_out[0 : T*D]
//   1) transpose+convert w1,w3 ([E,D,F] f32 -> [E,F,D] bf16) and
//      w2 ([E,F,D] f32 -> [E,D,F] bf16)  (B-side of WMMA wants N-major/K-contig)
//   2) RMSNorm + bf16 x + router logits + softmax-top2 dense weights we[T][E]
//   3) per expert e: GEMM1: h = silu(x@W1)*(x@W3)  (bf16 WMMA, f32 acc)
//                    GEMM2: out += we[:,e] * (h@W2)
// GEMMs double-buffer LDS panels and stage them with
// GLOBAL_LOAD_ASYNC_TO_LDS_B128 (ASYNCcnt) when the builtin is available.
// ---------------------------------------------------------------------------

#define T_TOK 8192
#define DIM   1024
#define FFN   4096
#define NEXP  8
#define LDP   72          // LDS panel row stride in bf16 elems (144B, 16B-aligned, bank-safe)

typedef __attribute__((ext_vector_type(16))) __bf16 v16bf;
typedef __attribute__((ext_vector_type(8)))  __bf16 v8bf;
typedef __attribute__((ext_vector_type(4)))  __bf16 v4bf;
typedef __attribute__((ext_vector_type(8)))  float  v8f;
typedef __attribute__((ext_vector_type(4)))  float  v4f;

#if defined(__has_builtin)
#if __has_builtin(__builtin_amdgcn_global_load_async_to_lds_b128) && \
    __has_builtin(__builtin_amdgcn_s_wait_asynccnt)
#define HAVE_ASYNC_LDS 1
#endif
#endif

#ifdef HAVE_ASYNC_LDS
// Builtin signature (from hipcc diagnostic): params are 128-bit int-vector
// pointers with HIP address-space qualifiers (AS1 global, AS3 LDS).
typedef int v4i __attribute__((__vector_size__(4 * sizeof(int))));
typedef __attribute__((address_space(1))) v4i as1_v4i;
typedef __attribute__((address_space(3))) v4i as3_v4i;
#endif

// ---------------- WMMA helpers ----------------

__device__ inline v8f wmma_bf16(v16bf a, v16bf b, v8f c) {
  // D = A(16x32 bf16) * B(32x16 bf16) + C(16x16 f32)
  return __builtin_amdgcn_wmma_f32_16x16x32_bf16(
      /*neg_a=*/false, a, /*neg_b=*/false, b,
      /*c_mod=*/(short)0, c, /*reuse_a=*/false, /*reuse_b=*/false);
}

// A fragment (16x32, M=token rows): lane<16 -> row=lane, K0..7 + K16..23;
// lane>=16 -> row=lane-16, K8..15 + K24..31.  (ISA 7.12.2 16-bit A table)
__device__ inline v16bf lds_load_afrag(const __bf16* p, int row0, int k0, int lane) {
  const int half = lane >> 4;
  const __bf16* base = p + (row0 + (lane & 15)) * LDP + k0 + half * 8;
  union { v16bf v; v8bf h[2]; } u;
  u.h[0] = *(const v8bf*)(base);
  u.h[1] = *(const v8bf*)(base + 16);
  return u.v;
}

// B fragment (32x16, N=cols): panel stores W^T rows (n-major, K contiguous).
// lane<16 -> col=lane, K0..7 + K16..23 ; lane>=16 -> col=lane-16, K8..15 + K24..31.
__device__ inline v16bf lds_load_bfrag(const __bf16* p, int col0, int k0, int lane) {
  const int half = lane >> 4;
  const __bf16* base = p + (col0 + (lane & 15)) * LDP + k0 + half * 8;
  union { v16bf v; v8bf h[2]; } u;
  u.h[0] = *(const v8bf*)(base);
  u.h[1] = *(const v8bf*)(base + 16);
  return u.v;
}

// Synchronous 128x64 panel copy (fallback path).
__device__ inline void copy_panel_sync(__bf16* dst, const __bf16* src, size_t srcStride, int tid) {
  const int r  = tid >> 3;          // 0..31
  const int c8 = (tid & 7) << 3;    // 0,8,...,56
  #pragma unroll
  for (int i = 0; i < 4; ++i) {
    *(v8bf*)(dst + (size_t)(r + 32 * i) * LDP + c8) =
        *(const v8bf*)(src + (size_t)(r + 32 * i) * srcStride + c8);
  }
}

// Async 128x64 panel copy: GLOBAL_LOAD_ASYNC_TO_LDS_B128 (4 issues/wave,
// ASYNCcnt += 4 per wave per panel). Falls back to sync copy if unavailable.
__device__ inline void copy_panel(__bf16* dst, const __bf16* src, size_t srcStride, int tid) {
#ifdef HAVE_ASYNC_LDS
  const int r  = tid >> 3;
  const int c8 = (tid & 7) << 3;
  #pragma unroll
  for (int i = 0; i < 4; ++i) {
    // flat->LDS: LDS byte address is the low 32 bits of the generic pointer
    unsigned loff = (unsigned)(uintptr_t)(dst + (size_t)(r + 32 * i) * LDP + c8);
    as3_v4i* lp = (as3_v4i*)loff;
    as1_v4i* gp = (as1_v4i*)(uintptr_t)(src + (size_t)(r + 32 * i) * srcStride + c8);
    __builtin_amdgcn_global_load_async_to_lds_b128(gp, lp, /*offset=*/0, /*cpol=*/0);
  }
#else
  copy_panel_sync(dst, src, srcStride, tid);
#endif
}

// Wait until at most N of this wave's async-to-LDS ops remain outstanding.
template <int N>
__device__ inline void wait_async() {
#ifdef HAVE_ASYNC_LDS
  __builtin_amdgcn_s_wait_asynccnt(N);
#endif
}

// ---------------- Stage 0: zero the dense output ----------------

__global__ __launch_bounds__(256) void zero_out_kernel(float* __restrict__ p) {
  v4f z = {0.f, 0.f, 0.f, 0.f};
  ((v4f*)p)[(size_t)blockIdx.x * 256 + threadIdx.x] = z;
}

// ---------------- Stage 1: transpose + f32->bf16 convert ----------------
// src: [E][R][C] f32, dst: [E][C][R] bf16. grid (C/32, R/32, E), block (32,8).

__global__ __launch_bounds__(256) void transpose_convert_kernel(
    const float* __restrict__ src, __bf16* __restrict__ dst, int R, int C) {
  __shared__ float tile[32][33];
  const size_t eoff = (size_t)blockIdx.z * R * C;
  const float* s = src + eoff;
  __bf16* d = dst + eoff;
  const int tx = threadIdx.x, ty = threadIdx.y;
  const int c0 = blockIdx.x * 32, r0 = blockIdx.y * 32;
  #pragma unroll
  for (int j = 0; j < 32; j += 8)
    tile[ty + j][tx] = s[(size_t)(r0 + ty + j) * C + c0 + tx];
  __syncthreads();
  #pragma unroll
  for (int j = 0; j < 32; j += 8)
    d[(size_t)(c0 + ty + j) * R + r0 + tx] = (__bf16)tile[tx][ty + j];
}

// ---------------- Stage 2: RMSNorm + router + top-2 ----------------
// One block per token, 256 threads (4 channels each).

__global__ __launch_bounds__(256) void norm_router_kernel(
    const float* __restrict__ hs, const float* __restrict__ nw,
    const float* __restrict__ gate, __bf16* __restrict__ xb,
    float* __restrict__ we, float* __restrict__ logits_out) {
  const int t = blockIdx.x;
  const int tid = threadIdx.x;
  const v4f xv = ((const v4f*)(hs + (size_t)t * DIM))[tid];

  __shared__ float sred[256];
  sred[tid] = xv.x * xv.x + xv.y * xv.y + xv.z * xv.z + xv.w * xv.w;
  __syncthreads();
  #pragma unroll
  for (int s = 128; s > 0; s >>= 1) {
    if (tid < s) sred[tid] += sred[tid + s];
    __syncthreads();
  }
  __shared__ float s_rms;
  if (tid == 0) s_rms = rsqrtf(sred[0] * (1.f / (float)DIM) + 1e-6f);
  __syncthreads();
  const float rms = s_rms;

  const v4f nwv = ((const v4f*)nw)[tid];
  const float x0 = xv.x * rms * nwv.x;
  const float x1 = xv.y * rms * nwv.y;
  const float x2 = xv.z * rms * nwv.z;
  const float x3 = xv.w * rms * nwv.w;

  v4bf xo;
  xo.x = (__bf16)x0; xo.y = (__bf16)x1; xo.z = (__bf16)x2; xo.w = (__bf16)x3;
  ((v4bf*)(xb + (size_t)t * DIM))[tid] = xo;

  // router partials: 8 dot products over this thread's 4 channels
  float acc[NEXP];
  #pragma unroll
  for (int e = 0; e < NEXP; ++e) {
    const v4f g = ((const v4f*)(gate + (size_t)e * DIM))[tid];
    acc[e] = x0 * g.x + x1 * g.y + x2 * g.z + x3 * g.w;
  }
  // wave reduce (wave32), then cross-wave via LDS
  #pragma unroll
  for (int off = 16; off > 0; off >>= 1) {
    #pragma unroll
    for (int e = 0; e < NEXP; ++e) acc[e] += __shfl_down(acc[e], off, 32);
  }
  __shared__ float swred[NEXP][8];
  const int wid = tid >> 5, lane = tid & 31;
  if (lane == 0) {
    #pragma unroll
    for (int e = 0; e < NEXP; ++e) swred[e][wid] = acc[e];
  }
  __syncthreads();
  if (tid == 0) {
    float l[NEXP];
    #pragma unroll
    for (int e = 0; e < NEXP; ++e) {
      float s = 0.f;
      #pragma unroll
      for (int w = 0; w < 8; ++w) s += swred[e][w];
      l[e] = s;
      logits_out[(size_t)t * NEXP + e] = s;
    }
    // softmax
    float m = l[0];
    #pragma unroll
    for (int e = 1; e < NEXP; ++e) m = fmaxf(m, l[e]);
    float p[NEXP], psum = 0.f;
    #pragma unroll
    for (int e = 0; e < NEXP; ++e) { p[e] = __expf(l[e] - m); psum += p[e]; }
    const float inv = 1.f / psum;
    #pragma unroll
    for (int e = 0; e < NEXP; ++e) p[e] *= inv;
    // top-2 (stable: first index wins ties, matching lax.top_k)
    int i1 = 0;
    #pragma unroll
    for (int e = 1; e < NEXP; ++e) if (p[e] > p[i1]) i1 = e;
    int i2 = (i1 == 0) ? 1 : 0;
    #pragma unroll
    for (int e = 0; e < NEXP; ++e) if (e != i1 && p[e] > p[i2]) i2 = e;
    const float s2 = p[i1] + p[i2];
    #pragma unroll
    for (int e = 0; e < NEXP; ++e) {
      float w = 0.f;
      if (e == i1) w = p[i1] / s2;
      else if (e == i2) w = p[i2] / s2;
      we[(size_t)t * NEXP + e] = w;
    }
  }
}

// ---------------- Stage 3a: GEMM1 (SwiGLU up-projection) ----------------
// h[t,f] = silu(x@W1)[t,f] * (x@W3)[t,f]   for one expert.
// grid (FFN/128, T/128), 256 threads = 8 waves arranged 2(M) x 4(N).
// Double-buffered LDS panels, async global->LDS staging.

__global__ __launch_bounds__(256) void moe_gemm1_kernel(
    const __bf16* __restrict__ xb, const __bf16* __restrict__ w1T,
    const __bf16* __restrict__ w3T, const float* __restrict__ we_e,
    __bf16* __restrict__ h) {
  const int tid = threadIdx.x, lane = tid & 31, wid = tid >> 5;
  const int wm = wid >> 2, wn = wid & 3;
  const int tile_t = blockIdx.y * 128, tile_f = blockIdx.x * 128;

  __shared__ int s_any;
  if (tid == 0) s_any = 0;
  __syncthreads();
  if (tid < 128 && we_e[(size_t)(tile_t + tid) * NEXP] != 0.f) atomicOr(&s_any, 1);
  __syncthreads();
  if (!s_any) return;  // no token in this tile routed to this expert

  __shared__ __align__(16) __bf16 sa [2][128 * LDP];
  __shared__ __align__(16) __bf16 sb1[2][128 * LDP];
  __shared__ __align__(16) __bf16 sb3[2][128 * LDP];

  v8f acc1[4][2], acc3[4][2];
  #pragma unroll
  for (int i = 0; i < 4; ++i)
    #pragma unroll
    for (int j = 0; j < 2; ++j) {
      v8f z = {0.f, 0.f, 0.f, 0.f, 0.f, 0.f, 0.f, 0.f};
      acc1[i][j] = z; acc3[i][j] = z;
    }

  const __bf16* asrc  = xb  + (size_t)tile_t * DIM;
  const __bf16* b1src = w1T + (size_t)tile_f * DIM;
  const __bf16* b3src = w3T + (size_t)tile_f * DIM;

  // prologue: stage panel 0
  copy_panel(sa[0],  asrc,  DIM, tid);
  copy_panel(sb1[0], b1src, DIM, tid);
  copy_panel(sb3[0], b3src, DIM, tid);

  const int NP = DIM / 64;
  for (int i = 0; i < NP; ++i) {
    const int cur = i & 1, nxt = cur ^ 1;
    if (i + 1 < NP) {
      const int kp = (i + 1) * 64;
      copy_panel(sa[nxt],  asrc  + kp, DIM, tid);
      copy_panel(sb1[nxt], b1src + kp, DIM, tid);
      copy_panel(sb3[nxt], b3src + kp, DIM, tid);
      wait_async<12>();   // leave only the 12 just-issued ops in flight
    } else {
      wait_async<0>();
    }
    __syncthreads();       // all waves' panel-cur data visible in LDS
    #pragma unroll
    for (int ks = 0; ks < 64; ks += 32) {
      v16bf af[4], bf1[2], bf3[2];
      #pragma unroll
      for (int fm = 0; fm < 4; ++fm) af[fm] = lds_load_afrag(sa[cur], wm * 64 + fm * 16, ks, lane);
      #pragma unroll
      for (int fn = 0; fn < 2; ++fn) {
        bf1[fn] = lds_load_bfrag(sb1[cur], wn * 32 + fn * 16, ks, lane);
        bf3[fn] = lds_load_bfrag(sb3[cur], wn * 32 + fn * 16, ks, lane);
      }
      #pragma unroll
      for (int fm = 0; fm < 4; ++fm)
        #pragma unroll
        for (int fn = 0; fn < 2; ++fn) {
          acc1[fm][fn] = wmma_bf16(af[fm], bf1[fn], acc1[fm][fn]);
          acc3[fm][fn] = wmma_bf16(af[fm], bf3[fn], acc3[fm][fn]);
        }
    }
    __syncthreads();       // done reading cur before it is refilled next iter
  }

  // epilogue: SwiGLU, store bf16 h
  const int half = lane >> 4, nl = lane & 15;
  #pragma unroll
  for (int fm = 0; fm < 4; ++fm)
    #pragma unroll
    for (int fn = 0; fn < 2; ++fn) {
      const int m0 = tile_t + wm * 64 + fm * 16 + half * 8;
      const int n  = tile_f + wn * 32 + fn * 16 + nl;
      #pragma unroll
      for (int v = 0; v < 8; ++v) {
        const float p1 = acc1[fm][fn][v];
        const float p3 = acc3[fm][fn][v];
        const float sw = p1 / (1.f + __expf(-p1));   // silu
        h[(size_t)(m0 + v) * FFN + n] = (__bf16)(sw * p3);
      }
    }
}

// ---------------- Stage 3b: GEMM2 (down-projection, weighted accumulate) ----
// out[t,d] += we[t] * (h @ W2)[t,d]   for one expert.
// grid (DIM/128, T/128), 256 threads = 8 waves arranged 2(M) x 4(N).

__global__ __launch_bounds__(256) void moe_gemm2_kernel(
    const __bf16* __restrict__ h, const __bf16* __restrict__ w2T,
    const float* __restrict__ we_e, float* __restrict__ out) {
  const int tid = threadIdx.x, lane = tid & 31, wid = tid >> 5;
  const int wm = wid >> 2, wn = wid & 3;
  const int tile_t = blockIdx.y * 128, tile_d = blockIdx.x * 128;

  __shared__ int s_any;
  __shared__ float s_w[128];
  if (tid == 0) s_any = 0;
  __syncthreads();
  if (tid < 128) {
    const float w = we_e[(size_t)(tile_t + tid) * NEXP];
    s_w[tid] = w;
    if (w != 0.f) atomicOr(&s_any, 1);
  }
  __syncthreads();
  if (!s_any) return;

  __shared__ __align__(16) __bf16 sa[2][128 * LDP];
  __shared__ __align__(16) __bf16 sb[2][128 * LDP];

  v8f acc[4][2];
  #pragma unroll
  for (int i = 0; i < 4; ++i)
    #pragma unroll
    for (int j = 0; j < 2; ++j) {
      v8f z = {0.f, 0.f, 0.f, 0.f, 0.f, 0.f, 0.f, 0.f};
      acc[i][j] = z;
    }

  const __bf16* asrc = h   + (size_t)tile_t * FFN;
  const __bf16* bsrc = w2T + (size_t)tile_d * FFN;

  copy_panel(sa[0], asrc, FFN, tid);
  copy_panel(sb[0], bsrc, FFN, tid);

  const int NP = FFN / 64;
  for (int i = 0; i < NP; ++i) {
    const int cur = i & 1, nxt = cur ^ 1;
    if (i + 1 < NP) {
      const int kp = (i + 1) * 64;
      copy_panel(sa[nxt], asrc + kp, FFN, tid);
      copy_panel(sb[nxt], bsrc + kp, FFN, tid);
      wait_async<8>();
    } else {
      wait_async<0>();
    }
    __syncthreads();
    #pragma unroll
    for (int ks = 0; ks < 64; ks += 32) {
      v16bf af[4], bfr[2];
      #pragma unroll
      for (int fm = 0; fm < 4; ++fm) af[fm] = lds_load_afrag(sa[cur], wm * 64 + fm * 16, ks, lane);
      #pragma unroll
      for (int fn = 0; fn < 2; ++fn) bfr[fn] = lds_load_bfrag(sb[cur], wn * 32 + fn * 16, ks, lane);
      #pragma unroll
      for (int fm = 0; fm < 4; ++fm)
        #pragma unroll
        for (int fn = 0; fn < 2; ++fn)
          acc[fm][fn] = wmma_bf16(af[fm], bfr[fn], acc[fm][fn]);
    }
    __syncthreads();
  }

  const int half = lane >> 4, nl = lane & 15;
  #pragma unroll
  for (int fm = 0; fm < 4; ++fm)
    #pragma unroll
    for (int fn = 0; fn < 2; ++fn) {
      const int ml0 = wm * 64 + fm * 16 + half * 8;   // local token row
      const int n   = tile_d + wn * 32 + fn * 16 + nl;
      #pragma unroll
      for (int v = 0; v < 8; ++v) {
        const float w = s_w[ml0 + v];
        const size_t idx = (size_t)(tile_t + ml0 + v) * DIM + n;
        out[idx] += w * acc[fm][fn][v];
      }
    }
}

// ---------------- Host launch ----------------

extern "C" void kernel_launch(void* const* d_in, const int* in_sizes, int n_in,
                              void* d_out, int out_size, void* d_ws, size_t ws_size,
                              hipStream_t stream) {
  (void)in_sizes; (void)n_in; (void)out_size; (void)ws_size;
  const float* hs   = (const float*)d_in[0];
  const float* nw   = (const float*)d_in[1];
  const float* gate = (const float*)d_in[2];
  const float* w1   = (const float*)d_in[3];
  const float* w3   = (const float*)d_in[4];
  const float* w2   = (const float*)d_in[5];
  // d_in[6] = top_k (fixed 2 per reference)

  float* out    = (float*)d_out;                 // [T, D]
  float* logits = out + (size_t)T_TOK * DIM;     // [T, E]

  char* ws = (char*)d_ws;
  __bf16* xb   = (__bf16*)ws;  ws += (size_t)T_TOK * DIM * 2;
  float*  we   = (float*)ws;   ws += (size_t)T_TOK * NEXP * 4;
  __bf16* hbuf = (__bf16*)ws;  ws += (size_t)T_TOK * FFN * 2;
  __bf16* w1T  = (__bf16*)ws;  ws += (size_t)NEXP * DIM * FFN * 2;
  __bf16* w3T  = (__bf16*)ws;  ws += (size_t)NEXP * DIM * FFN * 2;
  __bf16* w2T  = (__bf16*)ws;  ws += (size_t)NEXP * DIM * FFN * 2;

  // 0) zero dense output region (T*D floats, 1024 per block via float4)
  zero_out_kernel<<<(size_t)T_TOK * DIM / 1024, 256, 0, stream>>>(out);

  // 1) weight transpose+convert: w1,w3: [D,F]->[F,D]; w2: [F,D]->[D,F]
  dim3 tb(32, 8);
  transpose_convert_kernel<<<dim3(FFN / 32, DIM / 32, NEXP), tb, 0, stream>>>(w1, w1T, DIM, FFN);
  transpose_convert_kernel<<<dim3(FFN / 32, DIM / 32, NEXP), tb, 0, stream>>>(w3, w3T, DIM, FFN);
  transpose_convert_kernel<<<dim3(DIM / 32, FFN / 32, NEXP), tb, 0, stream>>>(w2, w2T, FFN, DIM);

  // 2) RMSNorm + router
  norm_router_kernel<<<T_TOK, 256, 0, stream>>>(hs, nw, gate, xb, we, logits);

  // 3) expert loop (stream-ordered; weights stay L2-resident per expert)
  for (int e = 0; e < NEXP; ++e) {
    const size_t wo = (size_t)e * DIM * FFN;
    moe_gemm1_kernel<<<dim3(FFN / 128, T_TOK / 128), 256, 0, stream>>>(
        xb, w1T + wo, w3T + wo, we + e, hbuf);
    moe_gemm2_kernel<<<dim3(DIM / 128, T_TOK / 128), 256, 0, stream>>>(
        hbuf, w2T + wo, we + e, out);
  }
}
